// FakeNewsAttention_24893630447842
// MI455X (gfx1250) — matrix operand order repaired and checked by reading, not compiled, not run on hardware
//
#include <hip/hip_runtime.h>
#include <hip/hip_bf16.h>
#include <cstdint>

typedef __attribute__((ext_vector_type(16))) __bf16 bf16x16;
typedef __attribute__((ext_vector_type(8)))  float  f32x8;

#define N_USERS 2213
#define S_TOT   50000
#define SPAD    50048        // padded source count (multiple of 32, covers all split tails)
#define D       128
#define H       8
#define BATCH   1024
#define SPLITS  4
#define S_SPLIT 12512        // 4*12512 = 50048, multiple of 32
#define STEP    32
#define NSTEPS  (S_SPLIT / STEP)   // 391
#define LOG2E   1.4426950408889634f
#define NEGINF  (-3.0e38f)

// LDS strides (element units), padded to dodge bank conflicts, 16B-aligned rows
#define MS_STRIDE 136   // 128 + 8 bf16
#define MT_STRIDE 40    // 32 + 8 bf16
#define AT_STRIDE 36    // 32 + 4 f32
#define P_STRIDE  40    // 32 + 8 bf16

union BFrag { bf16x16 v; uint4 q[2]; unsigned u[8]; };

__device__ __forceinline__ unsigned short f2bf(float x) {
    unsigned u = __float_as_uint(x);
    unsigned r = 0x7FFFu + ((u >> 16) & 1u);   // round-to-nearest-even
    return (unsigned short)((u + r) >> 16);
}

// LDS aperture keeps the byte offset in addr[31:0]; truncation yields the DS address.
__device__ __forceinline__ unsigned lds_off(const void* p) {
    return (unsigned)(uintptr_t)p;
}

// Async copy global -> LDS (CDNA5, ASYNCcnt-tracked).
__device__ __forceinline__ void async_b128(unsigned lds, const void* gaddr) {
    asm volatile("global_load_async_to_lds_b128 %0, %1, off"
                 :: "v"(lds), "v"((unsigned long long)(uintptr_t)gaddr) : "memory");
}
__device__ __forceinline__ void async_b64(unsigned lds, const void* gaddr) {
    asm volatile("global_load_async_to_lds_b64 %0, %1, off"
                 :: "v"(lds), "v"((unsigned long long)(uintptr_t)gaddr) : "memory");
}
__device__ __forceinline__ void wait_async0() {
    asm volatile("s_wait_asynccnt 0" ::: "memory");
}

// ---------------- phase 0: gather Xu ----------------
__global__ __launch_bounds__(256) void prep_xu(const int* __restrict__ Xid,
                                               const float* __restrict__ Uemb,
                                               float* __restrict__ Xu) {
    int idx = blockIdx.x * 256 + threadIdx.x;        // BATCH*D
    int d = idx & (D - 1);
    int b = idx >> 7;
    Xu[idx] = Uemb[(size_t)Xid[b] * D + d];
}

// ---------------- phase 0: Y[h] = (Xu @ Wcm[h]) / sqrt(D), bf16 ----------------
__global__ __launch_bounds__(256) void prep_y(const float* __restrict__ Xu,
                                              const float* __restrict__ Wcm,
                                              unsigned short* __restrict__ Ybf) {
    int idx = blockIdx.x * 256 + threadIdx.x;        // H*BATCH*D
    int j = idx & (D - 1);
    int b = (idx >> 7) & (BATCH - 1);
    int h = idx >> 17;
    const float* w = Wcm + (size_t)h * D * D;
    const float* x = Xu + (size_t)b * D;
    float acc = 0.f;
    #pragma unroll 8
    for (int d = 0; d < D; ++d) acc = fmaf(x[d], w[(size_t)d * D + j], acc);
    Ybf[idx] = f2bf(acc * 0.088388347648318447f);    // 1/sqrt(128)
}

// ---------------- phase 0: cast M to bf16, row-major + transposed, zero-padded ----------------
__global__ __launch_bounds__(256) void prep_m(const float* __restrict__ Msrc,
                                              unsigned short* __restrict__ Mbf,
                                              unsigned short* __restrict__ MbT) {
    int idx = blockIdx.x * 256 + threadIdx.x;        // SPAD*D
    int d = idx & (D - 1);
    int s = idx >> 7;
    float v = (s < S_TOT) ? Msrc[(size_t)s * D + d] : 0.f;
    unsigned short bv = f2bf(v);
    Mbf[idx] = bv;
    MbT[(size_t)d * SPAD + s] = bv;
}

// ---------------- phase 1: fused flash-attention over sources ----------------
__global__ __launch_bounds__(256) void attn_phase1(
    const int* __restrict__ Xid, const float* __restrict__ A_us,
    const unsigned short* __restrict__ Ybf,
    const unsigned short* __restrict__ Mbf,
    const unsigned short* __restrict__ MbT,
    float* __restrict__ pacc, float* __restrict__ pmax, float* __restrict__ psum)
{
    __shared__ __align__(16) unsigned short Ms[2][32 * MS_STRIDE];   // M tile, [s][d]
    __shared__ __align__(16) unsigned short Mt[2][128 * MT_STRIDE];  // M tile transposed, [d][s]
    __shared__ __align__(16) float At[2][16 * AT_STRIDE];            // A tile, [b][s]
    __shared__ __align__(16) unsigned short Pshm[H][16 * P_STRIDE];  // per-wave p scratch, [b][s]

    const int t = threadIdx.x;
    const int btile = blockIdx.x;
    const int split = blockIdx.y;
    const int h = t >> 5;                 // wave id == head
    const int lane = t & 31;
    const int s_base_g = split * S_SPLIT;
    const int count = (s_base_g + S_SPLIT <= S_TOT) ? S_SPLIT : (S_TOT - s_base_g);

    // ---- cooperative-loader indices (all 256 threads) ----
    const int ms_row0 = (t * 2) >> 4,     ms_ch0 = (t * 2) & 15;
    const int ms_row1 = (t * 2 + 1) >> 4, ms_ch1 = (t * 2 + 1) & 15;
    const int mt_d0 = (t * 2) >> 2,       mt_c0 = (t * 2) & 3;
    const int mt_d1 = (t * 2 + 1) >> 2,   mt_c1 = (t * 2 + 1) & 3;
    const int a_b = t >> 4, a_s = (t & 15) * 2;
    const long a_row = (long)Xid[btile * 16 + a_b] * S_TOT;

    // async DMA of the next tile straight into LDS (bf16 M tiles are pre-padded,
    // so the copies are unconditional; A needs a guarded fallback only at the
    // very tail of the last split).
    auto loader = [&](int it, int buf) {
        const int s0g = s_base_g + it * STEP;
        async_b128(lds_off(&Ms[buf][ms_row0 * MS_STRIDE + ms_ch0 * 8]),
                   Mbf + (size_t)(s0g + ms_row0) * D + ms_ch0 * 8);
        async_b128(lds_off(&Ms[buf][ms_row1 * MS_STRIDE + ms_ch1 * 8]),
                   Mbf + (size_t)(s0g + ms_row1) * D + ms_ch1 * 8);
        async_b128(lds_off(&Mt[buf][mt_d0 * MT_STRIDE + mt_c0 * 8]),
                   MbT + (size_t)mt_d0 * SPAD + s0g + mt_c0 * 8);
        async_b128(lds_off(&Mt[buf][mt_d1 * MT_STRIDE + mt_c1 * 8]),
                   MbT + (size_t)mt_d1 * SPAD + s0g + mt_c1 * 8);
        const int sg = s0g + a_s;
        if (s0g + STEP <= S_TOT) {
            async_b64(lds_off(&At[buf][a_b * AT_STRIDE + a_s]), A_us + a_row + sg);
        } else {
            float a0 = (sg     < S_TOT) ? A_us[a_row + sg]     : 0.f;
            float a1 = (sg + 1 < S_TOT) ? A_us[a_row + sg + 1] : 0.f;
            At[buf][a_b * AT_STRIDE + a_s]     = a0;
            At[buf][a_b * AT_STRIDE + a_s + 1] = a1;
        }
    };

    // ---- per-wave WMMA lane mapping (wave32) ----
    const int col = lane & 15;            // N index: batch column within tile
    const int hi = lane >> 4;             // lane half
    const int bglob = btile * 16 + col;
    const int kpartB = hi * 16;           // unused; B operands also use kbA
    const int kbA = hi * 8;               // A-operand K sub-base
    const int mrow_off = hi * 8;          // C/D layout: M = v + 8*hi

    // Y fragments (B-operand, one per K-chunk of 32), held in registers
    BFrag yf[4];
    {
        const unsigned short* yrow = Ybf + (size_t)(h * BATCH + bglob) * D;
        #pragma unroll
        for (int c = 0; c < 4; ++c) {
            yf[c].q[0] = *(const uint4*)(yrow + c * 32 + kbA);
            yf[c].q[1] = *(const uint4*)(yrow + c * 32 + kbA + 16);
        }
    }

    const f32x8 vzero = {0.f, 0.f, 0.f, 0.f, 0.f, 0.f, 0.f, 0.f};
    f32x8 acc[8];
    #pragma unroll
    for (int i = 0; i < 8; ++i) acc[i] = vzero;
    float runmax = 0.f, runsum = 0.f;     // z >= 0 always, so 0 is a valid floor

    loader(0, 0);
    wait_async0();
    __syncthreads();

    for (int it = 0; it < NSTEPS; ++it) {
        const int cur = it & 1;
        const int s_loc0 = it * STEP;

        // overlap: DMA next tile into the other buffer while we compute
        if (it + 1 < NSTEPS) loader(it + 1, cur ^ 1);

        // ---- scores: Z[s,b] = M_tile @ Y^T, two 16x16 subtiles, K=128 in 4 chunks ----
        f32x8 z[2];
        #pragma unroll
        for (int sub = 0; sub < 2; ++sub) {
            const int srow = sub * 16 + col;
            BFrag af[4];
            #pragma unroll
            for (int c = 0; c < 4; ++c) {
                const unsigned short* mr = &Ms[cur][srow * MS_STRIDE + c * 32 + kbA];
                af[c].q[0] = *(const uint4*)(mr);
                af[c].q[1] = *(const uint4*)(mr + 16);
            }
            f32x8 c8 = vzero;
            #pragma unroll
            for (int c = 0; c < 4; ++c)
                c8 = __builtin_amdgcn_wmma_f32_16x16x32_bf16(
                        false, af[c].v, false, yf[c].v, (short)0, c8, false, false);
            z[sub] = c8;
        }

        // ---- relu, *A, tail mask, tile max ----
        float tmax = NEGINF;
        #pragma unroll
        for (int sub = 0; sub < 2; ++sub) {
            const float* arow = &At[cur][col * AT_STRIDE + sub * 16 + mrow_off];
            float4 a0 = *(const float4*)(arow);
            float4 a1 = *(const float4*)(arow + 4);
            float av[8] = {a0.x, a0.y, a0.z, a0.w, a1.x, a1.y, a1.z, a1.w};
            #pragma unroll
            for (int v = 0; v < 8; ++v) {
                float zz = fmaxf(z[sub][v], 0.f) * av[v];
                int sl = s_loc0 + sub * 16 + mrow_off + v;
                zz = (sl < count) ? zz : NEGINF;
                z[sub][v] = zz;
                tmax = fmaxf(tmax, zz);
            }
        }

        // ---- online softmax update (column b lives in lane pair {col, col+16}) ----
        tmax = fmaxf(tmax, __shfl_xor(tmax, 16, 32));
        const float newmax = fmaxf(runmax, tmax);
        const float corr = exp2f((runmax - newmax) * LOG2E);
        runmax = newmax;
        runsum *= corr;
        #pragma unroll
        for (int i = 0; i < 8; ++i) acc[i] *= corr;

        // ---- p = exp(z - max), bf16, through per-wave LDS scratch ----
        float ps = 0.f;
        #pragma unroll
        for (int sub = 0; sub < 2; ++sub) {
            unsigned short* prow = &Pshm[h][col * P_STRIDE + sub * 16 + mrow_off];
            #pragma unroll
            for (int v = 0; v < 8; v += 2) {
                float p0 = exp2f((z[sub][v]     - newmax) * LOG2E);
                float p1 = exp2f((z[sub][v + 1] - newmax) * LOG2E);
                ps += p0 + p1;
                unsigned pk = (unsigned)f2bf(p0) | ((unsigned)f2bf(p1) << 16);
                *(unsigned*)(prow + v) = pk;
            }
        }
        ps += __shfl_xor(ps, 16, 32);
        runsum += ps;

        // ---- reload p as B-operand (LDS per-wave ops are in-order) ----
        BFrag pf;
        {
            const unsigned short* pr = &Pshm[h][col * P_STRIDE + kbA];
            pf.q[0] = *(const uint4*)(pr);
            pf.q[1] = *(const uint4*)(pr + 16);
        }
        // ---- acc[e,b] += M^T @ p : 8 e-tiles, distance-1 operand prefetch ----
        {
            BFrag mf[2];
            {
                const unsigned short* mr = &Mt[cur][(col) * MT_STRIDE + kbA];
                mf[0].q[0] = *(const uint4*)(mr);
                mf[0].q[1] = *(const uint4*)(mr + 16);
            }
            #pragma unroll
            for (int et = 0; et < 8; ++et) {
                if (et + 1 < 8) {
                    const unsigned short* mr = &Mt[cur][((et + 1) * 16 + col) * MT_STRIDE + kbA];
                    mf[(et + 1) & 1].q[0] = *(const uint4*)(mr);
                    mf[(et + 1) & 1].q[1] = *(const uint4*)(mr + 16);
                }
                acc[et] = __builtin_amdgcn_wmma_f32_16x16x32_bf16(
                            false, mf[et & 1].v, false, pf.v, (short)0, acc[et], false, false);
            }
        }

        // wait for our async DMA of the next tile, then rendezvous
        wait_async0();
        __syncthreads();
    }

    // ---- write per-split partials (unnormalized) ----
    if (lane < 16) {
        size_t r = (size_t)(split * H + h) * BATCH + bglob;
        pmax[r] = runmax;
        psum[r] = runsum;
    }
    {
        float* ap = pacc + ((size_t)(split * H + h) * BATCH + bglob) * D;
        #pragma unroll
        for (int et = 0; et < 8; ++et)
            #pragma unroll
            for (int v = 0; v < 8; ++v)
                ap[et * 16 + mrow_off + v] = acc[et][v];
    }
}

// ---------------- phase 2: merge splits (flash combine) ----------------
__global__ __launch_bounds__(256) void attn_combine(
    const float* __restrict__ pacc, const float* __restrict__ pmax,
    const float* __restrict__ psum, float* __restrict__ mh)
{
    int idx = blockIdx.x * 256 + threadIdx.x;      // H*BATCH*D
    int e = idx & (D - 1);
    int b = (idx >> 7) & (BATCH - 1);
    int h = idx >> 17;
    float M = NEGINF;
    #pragma unroll
    for (int s = 0; s < SPLITS; ++s)
        M = fmaxf(M, pmax[(size_t)(s * H + h) * BATCH + b]);
    float den = 0.f, num = 0.f;
    #pragma unroll
    for (int s = 0; s < SPLITS; ++s) {
        size_t r = (size_t)(s * H + h) * BATCH + b;
        float w = exp2f((pmax[r] - M) * LOG2E);
        den += psum[r] * w;
        num += pacc[r * D + e] * w;
    }
    mh[(size_t)b * (H * D) + h * D + e] = num / den;
}

// ---------------- phase 3: m = concat @ W1, elu, residual ----------------
__global__ __launch_bounds__(256) void attn_final(
    const float* __restrict__ mh, const float* __restrict__ W1,
    const float* __restrict__ Xu, float* __restrict__ out)
{
    int idx = blockIdx.x * 256 + threadIdx.x;      // BATCH*D
    int j = idx & (D - 1);
    int b = idx >> 7;
    const float* row = mh + (size_t)b * (H * D);
    float s = 0.f;
    #pragma unroll 8
    for (int c = 0; c < H * D; ++c)
        s = fmaf(row[c], W1[(size_t)c * D + j], s);
    float e = (s > 0.f) ? s : (__expf(s) - 1.0f);
    out[idx] = e + Xu[idx];
}

extern "C" void kernel_launch(void* const* d_in, const int* in_sizes, int n_in,
                              void* d_out, int out_size, void* d_ws, size_t ws_size,
                              hipStream_t stream)
{
    const int*   Xid  = (const int*)d_in[0];
    const float* Uemb = (const float*)d_in[1];
    const float* Semb = (const float*)d_in[2];
    const float* Wcm  = (const float*)d_in[3];
    const float* W1   = (const float*)d_in[4];
    const float* A_us = (const float*)d_in[5];
    float* out = (float*)d_out;

    char* ws = (char*)d_ws;
    float* Xu            = (float*)ws;          ws += (size_t)BATCH * D * 4;
    unsigned short* Ybf  = (unsigned short*)ws; ws += (size_t)H * BATCH * D * 2;
    unsigned short* Mbf  = (unsigned short*)ws; ws += (size_t)SPAD * D * 2;
    unsigned short* MbT  = (unsigned short*)ws; ws += (size_t)SPAD * D * 2;
    float* pacc          = (float*)ws;          ws += (size_t)SPLITS * H * BATCH * D * 4;
    float* pmax          = (float*)ws;          ws += (size_t)SPLITS * H * BATCH * 4;
    float* psum          = (float*)ws;          ws += (size_t)SPLITS * H * BATCH * 4;
    float* mh            = (float*)ws;          ws += (size_t)BATCH * H * D * 4;

    prep_xu<<<(BATCH * D) / 256, 256, 0, stream>>>(Xid, Uemb, Xu);
    prep_y<<<(H * BATCH * D) / 256, 256, 0, stream>>>(Xu, Wcm, Ybf);
    prep_m<<<(SPAD * D) / 256, 256, 0, stream>>>(Semb, Mbf, MbT);
    attn_phase1<<<dim3(BATCH / 16, SPLITS), 256, 0, stream>>>(
        Xid, A_us, Ybf, Mbf, MbT, pacc, pmax, psum);
    attn_combine<<<(H * BATCH * D) / 256, 256, 0, stream>>>(pacc, pmax, psum, mh);
    attn_final<<<(BATCH * D) / 256, 256, 0, stream>>>(mh, W1, Xu, out);
}
